// PromptEncoder_39591008534797
// MI455X (gfx1250) — compile-verified, matrix operand order checked by reading, not run on hardware
//
#include <hip/hip_runtime.h>

typedef _Float16 v16h __attribute__((ext_vector_type(16)));
typedef _Float16 v8h  __attribute__((ext_vector_type(8)));
typedef float    v8f  __attribute__((ext_vector_type(8)));

#define WMMA_F16(a,b,c) __builtin_amdgcn_wmma_f32_16x16x32_f16(false,(a),false,(b),(short)0,(c),false,false)

__device__ __forceinline__ int iclampi(int v, int lo, int hi){ return v<lo?lo:(v>hi?hi:v); }

// ---------------------------------------------------------------------------
// 1. crop_and_resize : (B*P, 3, 128, 128) bilinear from selected source image
// ---------------------------------------------------------------------------
__global__ void crop_resize_kernel(const float* __restrict__ imgs,   // (B,S,3,512,512)
                                   const int*   __restrict__ srcidx, // (B,P)
                                   const float* __restrict__ boxes,  // (B*P,4)
                                   float* __restrict__ crop)         // (B*P,3,128,128)
{
  const int n = blockIdx.y;
  const int b = n >> 4;
  int pix = blockIdx.x * blockDim.x + threadIdx.x;
  if (pix >= 128*128) return;
  int y = pix >> 7, x = pix & 127;

  const float* bx = boxes + (size_t)n*4;
  int b0 = (int)rintf(bx[0]); int b1 = (int)rintf(bx[1]);
  int b2 = (int)rintf(bx[2]); int b3 = (int)rintf(bx[3]);
  int x1 = iclampi(b0, 0, 511);
  int y1 = iclampi(b1, 0, 511);
  int x2 = max(x1 + 1, min(b2, 512));
  int y2 = max(y1 + 1, min(b3, 512));

  float sy = (float)(y2 - y1) / 128.0f;
  float srcy = fmaxf(sy * ((float)y + 0.5f) - 0.5f, 0.0f);
  int i0y = min((int)floorf(srcy), y2 - y1 - 1);
  int i1y = min(i0y + 1, y2 - y1 - 1);
  float wy = fminf(fmaxf(srcy - (float)i0y, 0.0f), 1.0f);
  int iy0 = y1 + i0y, iy1 = y1 + i1y;

  float sx = (float)(x2 - x1) / 128.0f;
  float srcx = fmaxf(sx * ((float)x + 0.5f) - 0.5f, 0.0f);
  int i0x = min((int)floorf(srcx), x2 - x1 - 1);
  int i1x = min(i0x + 1, x2 - x1 - 1);
  float wx = fminf(fmaxf(srcx - (float)i0x, 0.0f), 1.0f);
  int ix0 = x1 + i0x, ix1 = x1 + i1x;

  int si = max(srcidx[n], 0);
  const float* im = imgs + ((size_t)(b*4 + si) * 3) * 512 * 512;
  #pragma unroll
  for (int c = 0; c < 3; ++c) {
    const float* ch = im + (size_t)c * 512 * 512;
    float v00 = ch[iy0*512 + ix0], v01 = ch[iy0*512 + ix1];
    float v10 = ch[iy1*512 + ix0], v11 = ch[iy1*512 + ix1];
    float v = v00*(1.f-wy)*(1.f-wx) + v01*(1.f-wy)*wx + v10*wy*(1.f-wx) + v11*wy*wx;
    crop[((size_t)n*3 + c)*16384 + pix] = v;
  }
}

// ---------------------------------------------------------------------------
// 2a. prepack conv weights to f16 in WMMA A-fragment order
//     layout: [ktile][mtile][lane(32)][16 halves]
// ---------------------------------------------------------------------------
__global__ void pack_weights_kernel(const float* __restrict__ W, // (Cout, Ktot)
                                    _Float16* __restrict__ Wpk,
                                    int Cout, int Ktot)
{
  int Mtiles = Cout >> 4;
  int Ktiles = (Ktot + 31) >> 5;
  int id = blockIdx.x * blockDim.x + threadIdx.x;   // (kt,mt,lane)
  if (id >= Ktiles * Mtiles * 32) return;
  int lane = id & 31;
  int mt = (id >> 5) % Mtiles;
  int g = lane >> 4, l = lane & 15;
  int kt = (id >> 5) / Mtiles;
  _Float16* dst = Wpk + (size_t)id * 16;
  #pragma unroll
  for (int i = 0; i < 16; ++i) {
    int k = kt*32 + 8*g + i + (i < 8 ? 0 : 8);
    float v = (k < Ktot) ? W[(size_t)(mt*16 + l) * Ktot + k] : 0.f;
    dst[i] = (_Float16)v;
  }
}

// ---------------------------------------------------------------------------
// 2b. conv3x3 s2 p1 + BN + ReLU, implicit GEMM.
//     Block = (Cout/16) waves, covers all Cout x 32 spatial columns of image n.
//     Per k-tile: 32-entry k->(ci,ky,kx) decode table hoisted to LDS, im2col
//     staged once per block (2 halves packed per b32 store), then per wave:
//     1 x 32B A load, 4 x ds_load_b128, 2 x WMMA.
// ---------------------------------------------------------------------------
#define BCOL_STRIDE 40   // halves per column (80B, conflict-free, 16B multiple)

__global__ __launch_bounds__(512) void conv_bn_relu_wmma(
    const float* __restrict__ in,       // (N, Cin, Hin, Win)
    const _Float16* __restrict__ Wpk,   // prepacked fragments
    const float* __restrict__ gam, const float* __restrict__ bet,
    const float* __restrict__ mu,  const float* __restrict__ var,
    float* __restrict__ out,            // (N, Cout, Hout, Wout)
    int Cin, int Cout, int Hin, int Win, int wlog)
{
  __shared__ __align__(16) _Float16 Bs[32 * BCOL_STRIDE];
  __shared__ int toff[32];   // ci*Hin*Win + (ky-1)*Win + (kx-1)
  __shared__ int tkxy[32];   // ky | kx<<8, or -1 if k >= Ktot
  const int Hout = Hin >> 1, Wout = Win >> 1;
  const int S = Hout * Wout;
  const int n  = blockIdx.x;
  const int nb = blockIdx.y * 32;
  const int tid  = threadIdx.x;
  const int wave = tid >> 5, lane = tid & 31;
  const int g = lane >> 4, l = lane & 15;
  const int Ktot = Cin * 9;
  const int Ktiles = (Ktot + 31) >> 5;
  const int Mtiles = Cout >> 4;
  const float* inN = in + (size_t)n * Cin * Hin * Win;

  v8f acc0 = {}, acc1 = {};
  for (int kt = 0; kt < Ktiles; ++kt) {
    __syncthreads();                       // prev-iter LDS reads done
    if (tid < 32) {                        // hoisted k -> (ci,ky,kx) decode
      int k = kt*32 + tid;
      int ci = k / 9, r9 = k % 9;
      int ky = r9 / 3, kx = r9 % 3;
      toff[tid] = ci * Hin * Win + (ky - 1) * Win + (kx - 1);
      tkxy[tid] = (k < Ktot) ? (ky | (kx << 8)) : -1;
    }
    __syncthreads();                       // table visible
    // cooperative im2col fill: 32 cols x 32 k, 2 halves per b32 store
    for (int v = tid; v < 512; v += blockDim.x) {
      int col = v >> 4, kp = (v & 15) << 1;
      int s = nb + col;
      int oy = s >> wlog, ox = s & (Wout - 1);
      int ibase = (oy * 2) * Win + ox * 2;
      unsigned int pack = 0;
      #pragma unroll
      for (int u = 0; u < 2; ++u) {
        int kk = kp + u;
        int t = tkxy[kk];
        float bv = 0.f;
        if (t >= 0) {
          int ky = t & 0xff, kx = t >> 8;
          int iy = oy*2 - 1 + ky, ix = ox*2 - 1 + kx;
          if (iy >= 0 && iy < Hin && ix >= 0 && ix < Win)
            bv = inN[toff[kk] + ibase];
        }
        unsigned short us = __builtin_bit_cast(unsigned short, (_Float16)bv);
        pack |= (unsigned int)us << (16*u);
      }
      *(unsigned int*)(Bs + col * BCOL_STRIDE + kp) = pack;
    }
    __syncthreads();                       // Bs visible
    // A fragment (prepacked, contiguous 32B per lane)
    v16h a = *(const v16h*)(Wpk + (((size_t)kt * Mtiles + wave) * 32 + lane) * 16);
    // B fragments from LDS: cols [0..15] and [16..31]
    const _Float16* bp0 = Bs + l * BCOL_STRIDE + 8*g;
    const _Float16* bp1 = Bs + (16 + l) * BCOL_STRIDE + 8*g;
    v8h x0 = *(const v8h*)bp0;
    v8h x1 = *(const v8h*)(bp0 + 16);
    v8h y0 = *(const v8h*)bp1;
    v8h y1 = *(const v8h*)(bp1 + 16);
    v16h b0 = __builtin_shufflevector(x0, x1, 0,1,2,3,4,5,6,7,8,9,10,11,12,13,14,15);
    v16h b1 = __builtin_shufflevector(y0, y1, 0,1,2,3,4,5,6,7,8,9,10,11,12,13,14,15);
    acc0 = WMMA_F16(a, b0, acc0);
    acc1 = WMMA_F16(a, b1, acc1);
  }
  // BN + ReLU epilogue
  #pragma unroll
  for (int r = 0; r < 8; ++r) {
    int co = wave*16 + r + 8*g;
    float inv = gam[co] * rsqrtf(var[co] + 1e-5f);
    float sh  = bet[co] - mu[co] * inv;
    out[((size_t)n * Cout + co) * S + nb + l]      = fmaxf(acc0[r]*inv + sh, 0.f);
    out[((size_t)n * Cout + co) * S + nb + 16 + l] = fmaxf(acc1[r]*inv + sh, 0.f);
  }
}

// ---------------------------------------------------------------------------
// 3. crop_global = mean over 64 spatial of crop_feat
// ---------------------------------------------------------------------------
__global__ void crop_global_kernel(const float* __restrict__ feat, // (N,256,64)
                                   float* __restrict__ cg)         // (N,256)
{
  int id = blockIdx.x * blockDim.x + threadIdx.x;
  if (id >= 32*256) return;
  const float* f = feat + (size_t)id * 64;
  float s = 0.f;
  #pragma unroll 8
  for (int t = 0; t < 64; ++t) s += f[t];
  cg[id] = s * (1.0f/64.0f);
}

// ---------------------------------------------------------------------------
// 4. ROI mean pool over p3/p4/p5, averaged -> support_global  (block per (n,c))
// ---------------------------------------------------------------------------
__global__ __launch_bounds__(64) void roi_support_kernel(
    const float* __restrict__ p3, const float* __restrict__ p4,
    const float* __restrict__ p5, const float* __restrict__ boxes,
    float* __restrict__ sg)
{
  int id = blockIdx.x;
  int n = id >> 8, c = id & 255;
  const float* bx = boxes + (size_t)n*4;
  float acc = 0.f;
  #pragma unroll
  for (int lvl = 0; lvl < 3; ++lvl) {
    int F = (lvl == 0) ? 64 : (lvl == 1 ? 32 : 16);
    const float* f = ((lvl == 0) ? p3 : (lvl == 1) ? p4 : p5)
                     + ((size_t)n * 256 + c) * F * F;
    float s = (float)F / 512.0f;
    int fx1 = iclampi((int)rintf(bx[0]*s), 0, F-1);
    int fy1 = iclampi((int)rintf(bx[1]*s), 0, F-1);
    int fx2 = max(fx1 + 1, min((int)rintf(bx[2]*s), F));
    int fy2 = max(fy1 + 1, min((int)rintf(bx[3]*s), F));
    int w = fx2 - fx1, h = fy2 - fy1;
    int wh = w * h;
    float ls = 0.f;
    for (int j = threadIdx.x; j < wh; j += 64) {
      int j2 = j + 64;
      if (j2 < wh)
        __builtin_prefetch(&f[(fy1 + j2 / w) * F + (fx1 + j2 % w)], 0, 1);
      ls += f[(fy1 + j / w) * F + (fx1 + j % w)];
    }
    acc += ls / ((float)wh * 3.0f);
  }
  __shared__ float red[64];
  red[threadIdx.x] = acc;
  __syncthreads();
  for (int off = 32; off > 0; off >>= 1) {
    if (threadIdx.x < off) red[threadIdx.x] += red[threadIdx.x + off];
    __syncthreads();
  }
  if (threadIdx.x == 0) sg[id] = red[0];
}

// ---------------------------------------------------------------------------
// 5. Generic WMMA linear: Y[r,o] = sum_k X[r,k]*W[o,k] + b[o]  (opt ReLU)
// ---------------------------------------------------------------------------
__global__ __launch_bounds__(32) void linear_wmma(
    const float* __restrict__ X, const float* __restrict__ W,
    const float* __restrict__ bias, float* __restrict__ Y,
    int R, int In, int Out, int relu)
{
  int ob = blockIdx.x * 16;
  int rb = blockIdx.y * 16;
  int lane = threadIdx.x, g = lane >> 4, l = lane & 15;
  v8f acc = {};
  for (int k0 = 0; k0 < In; k0 += 32) {
    v16h a, bfr;
    #pragma unroll
    for (int i = 0; i < 16; ++i) {
      int k = k0 + 8*g + i + (i < 8 ? 0 : 8);
      float av = 0.f, bv = 0.f;
      if (k < In) {
        av = W[(size_t)(ob + l) * In + k];
        int row = rb + l;
        if (row < R) bv = X[(size_t)row * In + k];
      }
      a[i] = (_Float16)av;
      bfr[i] = (_Float16)bv;
    }
    acc = WMMA_F16(a, bfr, acc);
  }
  int row = rb + l;
  if (row < R) {
    #pragma unroll
    for (int r = 0; r < 8; ++r) {
      int o = ob + r + 8*g;
      float v = acc[r] + bias[o];
      if (relu) v = fmaxf(v, 0.f);
      Y[(size_t)row * Out + o] = v;
    }
  }
}

// ---------------------------------------------------------------------------
// 6. local_proj (1x1 conv) + bias + slot-proj add + instance mask -> LT(N,T,C)
// ---------------------------------------------------------------------------
__global__ __launch_bounds__(32) void local_proj_wmma(
    const float* __restrict__ feat,    // (N,256,64)
    const float* __restrict__ Wp,      // (256,256)
    const float* __restrict__ bp,      // (256)
    const float* __restrict__ slotadd, // (N,256)
    const int*   __restrict__ imask,   // (N)
    float* __restrict__ LT)            // (N,64,256)
{
  int n = blockIdx.x;
  int cb = blockIdx.y * 16;
  int tb = blockIdx.z * 16;
  int lane = threadIdx.x, g = lane >> 4, l = lane & 15;
  const float* f = feat + (size_t)n * 256 * 64;
  v8f acc = {};
  for (int k0 = 0; k0 < 256; k0 += 32) {
    v16h a, bfr;
    #pragma unroll
    for (int i = 0; i < 16; ++i) {
      int k = k0 + 8*g + i + (i < 8 ? 0 : 8);
      a[i]   = (_Float16)Wp[(size_t)(cb + l) * 256 + k];
      bfr[i] = (_Float16)f[(size_t)k * 64 + tb + l];
    }
    acc = WMMA_F16(a, bfr, acc);
  }
  float m = imask[n] ? 1.f : 0.f;
  int t = tb + l;
  #pragma unroll
  for (int r = 0; r < 8; ++r) {
    int c = cb + r + 8*g;
    LT[((size_t)n * 64 + t) * 256 + c] = (acc[r] + bp[c] + slotadd[(size_t)n*256 + c]) * m;
  }
}

// ---------------------------------------------------------------------------
// 7. small elementwise glue kernels
// ---------------------------------------------------------------------------
__global__ void build_xcat_kernel(const float* __restrict__ cg,
                                  const float* __restrict__ sg,
                                  const float* __restrict__ boxes,
                                  float* __restrict__ xcat)  // (32,516)
{
  int id = blockIdx.x * blockDim.x + threadIdx.x;
  if (id >= 32*516) return;
  int n = id / 516, j = id % 516;
  float v;
  if (j < 256)      v = cg[n*256 + j];
  else if (j < 512) v = sg[n*256 + (j - 256)];
  else              v = boxes[n*4 + (j - 512)] * (1.0f/512.0f);
  xcat[id] = v;
}

__global__ void build_xslot_kernel(const int* __restrict__ cls,
                                   const float* __restrict__ slot_emb, // (8,256)
                                   float* __restrict__ xslot)          // (32,256)
{
  int id = blockIdx.x * blockDim.x + threadIdx.x;
  if (id >= 32*256) return;
  int n = id >> 8, c = id & 255;
  int ci = iclampi(cls[n], 0, 7);
  xslot[id] = slot_emb[ci*256 + c];
}

__global__ void add_vec_kernel(const float* __restrict__ a,
                               const float* __restrict__ b,
                               float* __restrict__ y, int n)
{
  int id = blockIdx.x * blockDim.x + threadIdx.x;
  if (id < n) y[id] = a[id] + b[id];
}

__global__ void mask_rows_kernel(float* __restrict__ x, const int* __restrict__ imask)
{
  int id = blockIdx.x * blockDim.x + threadIdx.x;
  if (id >= 32*256) return;
  if (!imask[id >> 8]) x[id] = 0.f;
}

// ---------------------------------------------------------------------------
// 8. class prototypes / class local einsums (16-deep sums over P)
// ---------------------------------------------------------------------------
__global__ void class_proto_kernel(const float* __restrict__ ig,  // (32,256) masked
                                   const int* __restrict__ cls, const int* __restrict__ imask,
                                   const int* __restrict__ pcm,
                                   float* __restrict__ proto)     // (2,8,256)
{
  int id = blockIdx.x * blockDim.x + threadIdx.x;
  if (id >= 2*8*256) return;
  int c = id & 255, bk = id >> 8, b = bk >> 3, k = bk & 7;
  float s = 0.f, cnt = 0.f;
  #pragma unroll
  for (int p = 0; p < 16; ++p) {
    int n = b*16 + p;
    float a = (iclampi(cls[n],0,7) == k && imask[n]) ? 1.f : 0.f;
    cnt += a;
    s += a * ig[(size_t)n*256 + c];
  }
  float pm = pcm[b*8 + k] ? 1.f : 0.f;
  proto[id] = s / fmaxf(cnt, 1.f) * pm;
}

__global__ void class_local_kernel(const float* __restrict__ LT,  // (32,64,256) masked
                                   const int* __restrict__ cls, const int* __restrict__ imask,
                                   const int* __restrict__ pcm,
                                   float* __restrict__ cl)        // (2,8,64,256)
{
  int id = blockIdx.x * blockDim.x + threadIdx.x;
  if (id >= 2*8*64*256) return;
  int c = id & 255, t = (id >> 8) & 63, k = (id >> 14) & 7, b = id >> 17;
  float s = 0.f, cnt = 0.f;
  #pragma unroll
  for (int p = 0; p < 16; ++p) {
    int n = b*16 + p;
    float a = (iclampi(cls[n],0,7) == k && imask[n]) ? 1.f : 0.f;
    cnt += a;
    s += a * LT[((size_t)n*64 + t)*256 + c];
  }
  float pm = pcm[b*8 + k] ? 1.f : 0.f;
  cl[id] = s / fmaxf(cnt, 1.f) * pm;
}

// ---------------------------------------------------------------------------
// 9. assemble the tuple output
// ---------------------------------------------------------------------------
__global__ void assemble_kernel(const float* __restrict__ cl,     // (2,512,256)
                                const float* __restrict__ LT,     // (2,1024,256)
                                const float* __restrict__ proto,  // (2,8,256)
                                const float* __restrict__ ig,     // (32,256)
                                const float* __restrict__ sc0, const float* __restrict__ sc1,
                                const float* __restrict__ sc2,
                                const float* __restrict__ si0, const float* __restrict__ si1,
                                const float* __restrict__ si2,
                                const int* __restrict__ imask, const int* __restrict__ pcm,
                                const int* __restrict__ cls,
                                float* __restrict__ out, int out_size)
{
  const int C = 256, PER_B = 1632 * C;                 // 417792
  const int MAIN = 2 * PER_B;                          // 835584
  const int TOTAL = MAIN + 2*1536 + 16 + 32;           // 838704
  int id = blockIdx.x * blockDim.x + threadIdx.x;
  if (id >= TOTAL || id >= out_size) return;
  float v;
  if (id < MAIN) {
    int b = id / PER_B, rem = id % PER_B;
    int row = rem / C, c = rem % C;
    if (row < 512)        v = cl[(size_t)b*512*256 + row*256 + c];
    else if (row < 1536)  v = LT[(size_t)b*1024*256 + (row-512)*256 + c];
    else if (row < 1544)  v = proto[(size_t)(b*8 + (row-1536))*256 + c];
    else if (row < 1560)  v = ig[(size_t)(b*16 + (row-1544))*256 + c];
    else if (row < 1584) {
      int which = (row - 1560) >> 3, kk = (row - 1560) & 7;
      const float* sc = which == 0 ? sc0 : (which == 1 ? sc1 : sc2);
      v = sc[(size_t)(b*8 + kk)*256 + c] * (pcm[b*8 + kk] ? 1.f : 0.f);
    } else {
      int which = (row - 1584) >> 4, pp = (row - 1584) & 15;
      const float* si = which == 0 ? si0 : (which == 1 ? si1 : si2);
      v = si[(size_t)(b*16 + pp)*256 + c] * (imask[b*16 + pp] ? 1.f : 0.f);
    }
  } else {
    int j = id - MAIN;
    if (j < 3072) {                          // memory_mask (2,1536)
      int b = j / 1536, r = j % 1536;
      if (r < 512) v = pcm[b*8 + (r >> 6)] ? 1.f : 0.f;
      else         v = imask[b*16 + ((r - 512) >> 6)] ? 1.f : 0.f;
    } else if (j < 3088) {                   // padded_class_mask (2,8)
      v = pcm[j - 3072] ? 1.f : 0.f;
    } else {                                 // instance_class_indices (2,16)
      int n = j - 3088;
      v = imask[n] ? (float)iclampi(cls[n], 0, 7) : -1.0f;
    }
  }
  out[id] = v;
}

// ---------------------------------------------------------------------------
extern "C" void kernel_launch(void* const* d_in, const int* in_sizes, int n_in,
                              void* d_out, int out_size, void* d_ws, size_t ws_size,
                              hipStream_t stream) {
  (void)in_sizes; (void)n_in; (void)ws_size;
  const float* imgs  = (const float*)d_in[0];
  const int*   srcix = (const int*)d_in[1];
  const float* boxes = (const float*)d_in[2];
  const float* p3    = (const float*)d_in[3];
  const float* p4    = (const float*)d_in[4];
  const float* p5    = (const float*)d_in[5];
  const int*   cls   = (const int*)d_in[6];
  const int*   imask = (const int*)d_in[7];
  const int*   pcm   = (const int*)d_in[8];
  // params (dict insertion order)
  const float* cew[4] = {(const float*)d_in[9],  (const float*)d_in[14],
                         (const float*)d_in[19], (const float*)d_in[24]};
  const float* ceg[4] = {(const float*)d_in[10], (const float*)d_in[15],
                         (const float*)d_in[20], (const float*)d_in[25]};
  const float* ceb[4] = {(const float*)d_in[11], (const float*)d_in[16],
                         (const float*)d_in[21], (const float*)d_in[26]};
  const float* cem[4] = {(const float*)d_in[12], (const float*)d_in[17],
                         (const float*)d_in[22], (const float*)d_in[27]};
  const float* cev[4] = {(const float*)d_in[13], (const float*)d_in[18],
                         (const float*)d_in[23], (const float*)d_in[28]};
  const float* lpW = (const float*)d_in[29]; const float* lpB = (const float*)d_in[30];
  const float* gpW = (const float*)d_in[31]; const float* gpB = (const float*)d_in[32];
  const float* slotE = (const float*)d_in[33];
  const float* lsW = (const float*)d_in[34]; const float* lsB = (const float*)d_in[35];
  const float* spW[3] = {(const float*)d_in[36], (const float*)d_in[38], (const float*)d_in[40]};
  const float* spB[3] = {(const float*)d_in[37], (const float*)d_in[39], (const float*)d_in[41]};
  const float* l1W = (const float*)d_in[42]; const float* l1B = (const float*)d_in[43];
  const float* l2W = (const float*)d_in[44]; const float* l2B = (const float*)d_in[45];

  float* ws = (float*)d_ws;
  float* crop  = ws;                       // 1,572,864  (reused for conv3 out)
  float* c1    = ws + 1572864;             // 6,291,456
  float* c2    = ws + 7864320;             // 3,145,728
  float* c3    = crop;                     // reuse
  float* c4    = ws + 11010048;            //   524,288
  float* LT    = ws + 11534336;            //   524,288
  float* cg    = ws + 12058624;
  float* sg    = ws + 12066816;
  float* xcat  = ws + 12075008;
  float* pv    = ws + 12091520;
  float* xslot = ws + 12099712;
  float* sadd  = ws + 12107904;
  float* rin   = ws + 12116096;
  float* hid   = ws + 12124288;
  float* ig    = ws + 12140672;
  float* proto = ws + 12148864;
  float* clb   = ws + 12152960;            //   262,144
  float* sc[3] = {ws + 12415104, ws + 12419200, ws + 12423296};
  float* si[3] = {ws + 12427392, ws + 12435584, ws + 12443776};
  // prepacked f16 conv weights (32B-aligned offsets)
  _Float16* wp1 = (_Float16*)(ws + 12451968);   //   1,536 halves
  _Float16* wp2 = (_Float16*)(ws + 12452736);   //  43,008 halves
  _Float16* wp3 = (_Float16*)(ws + 12474240);   // 165,888 halves
  _Float16* wp4 = (_Float16*)(ws + 12557184);   // 442,368 halves

  // 0. prepack conv weights to WMMA fragment layout (f16)
  pack_weights_kernel<<<(1*3*32   + 255)/256, 256, 0, stream>>>(cew[0], wp1,  48,   27);
  pack_weights_kernel<<<(14*6*32  + 255)/256, 256, 0, stream>>>(cew[1], wp2,  96,  432);
  pack_weights_kernel<<<(27*12*32 + 255)/256, 256, 0, stream>>>(cew[2], wp3, 192,  864);
  pack_weights_kernel<<<(54*16*32 + 255)/256, 256, 0, stream>>>(cew[3], wp4, 256, 1728);
  // 1. crop + resize
  crop_resize_kernel<<<dim3(64, 32), 256, 0, stream>>>(imgs, srcix, boxes, crop);
  // 2. conv stack (WMMA, LDS-staged im2col)
  conv_bn_relu_wmma<<<dim3(32,128),  96, 0, stream>>>(crop, wp1, ceg[0], ceb[0], cem[0], cev[0], c1,   3,  48, 128, 128, 6);
  conv_bn_relu_wmma<<<dim3(32, 32), 192, 0, stream>>>(c1,   wp2, ceg[1], ceb[1], cem[1], cev[1], c2,  48,  96,  64,  64, 5);
  conv_bn_relu_wmma<<<dim3(32,  8), 384, 0, stream>>>(c2,   wp3, ceg[2], ceb[2], cem[2], cev[2], c3,  96, 192,  32,  32, 4);
  conv_bn_relu_wmma<<<dim3(32,  2), 512, 0, stream>>>(c3,   wp4, ceg[3], ceb[3], cem[3], cev[3], c4, 192, 256,  16,  16, 3);
  // 3. crop_global mean
  crop_global_kernel<<<(32*256 + 255)/256, 256, 0, stream>>>(c4, cg);
  // 4. ROI support pooling (dominant HBM traffic)
  roi_support_kernel<<<32*256, 64, 0, stream>>>(p3, p4, p5, boxes, sg);
  // 5. prompt_vec = global_proj([cg | sg | box_norm])
  build_xcat_kernel<<<(32*516 + 255)/256, 256, 0, stream>>>(cg, sg, boxes, xcat);
  linear_wmma<<<dim3(16, 2), 32, 0, stream>>>(xcat, gpW, gpB, pv, 32, 516, 256, 0);
  // 6. slot embed gather + local_slot_proj
  build_xslot_kernel<<<(32*256 + 255)/256, 256, 0, stream>>>(cls, slotE, xslot);
  linear_wmma<<<dim3(16, 2), 32, 0, stream>>>(xslot, lsW, lsB, sadd, 32, 256, 256, 0);
  // 7. refine MLP: ig = l2(relu(l1(pv + slot)))
  add_vec_kernel<<<(32*256 + 255)/256, 256, 0, stream>>>(pv, xslot, rin, 32*256);
  linear_wmma<<<dim3(32, 2), 32, 0, stream>>>(rin, l1W, l1B, hid, 32, 256, 512, 1);
  linear_wmma<<<dim3(16, 2), 32, 0, stream>>>(hid, l2W, l2B, ig, 32, 512, 256, 0);
  mask_rows_kernel<<<(32*256 + 255)/256, 256, 0, stream>>>(ig, imask);
  // 8. local tokens (WMMA 1x1 conv + slot add + mask)
  local_proj_wmma<<<dim3(32, 16, 4), 32, 0, stream>>>(c4, lpW, lpB, sadd, imask, LT);
  // 9. class prototypes + class local
  class_proto_kernel<<<(2*8*256 + 255)/256, 256, 0, stream>>>(ig, cls, imask, pcm, proto);
  class_local_kernel<<<(2*8*64*256 + 255)/256, 256, 0, stream>>>(LT, cls, imask, pcm, clb);
  // 10. scale projections
  for (int i = 0; i < 3; ++i) {
    linear_wmma<<<dim3(16, 1), 32, 0, stream>>>(proto, spW[i], spB[i], sc[i], 16, 256, 256, 0);
    linear_wmma<<<dim3(16, 2), 32, 0, stream>>>(ig,    spW[i], spB[i], si[i], 32, 256, 256, 0);
  }
  // 11. assemble tuple output
  const int TOTAL = 838704;
  assemble_kernel<<<(TOTAL + 255)/256, 256, 0, stream>>>(
      clb, LT, proto, ig, sc[0], sc[1], sc[2], si[0], si[1], si[2],
      imask, pcm, cls, (float*)d_out, out_size);
}
